// _EdgeConvLayer_5583457485518
// MI455X (gfx1250) — matrix operand order repaired and checked by reading, not compiled, not run on hardware
//
#include <hip/hip_runtime.h>
#include <stdint.h>

#define DFEAT 128
#define EFEAT 16
#define KPAD  160   // 144 concat dims padded to 5 chunks of K=32
#define WAVES 8     // 256-thread block = 8 wave32

typedef __attribute__((ext_vector_type(16))) _Float16 v16h;
typedef __attribute__((ext_vector_type(8)))  _Float16 v8h;
typedef __attribute__((ext_vector_type(8)))  float    v8f;
typedef __attribute__((ext_vector_type(4)))  float    v4f;

__device__ __forceinline__ v8f wmma_f16(v16h a, v16h b, v8f c) {
  // D = A(16x32 f16) * B(32x16 f16) + C(16x16 f32)
  return __builtin_amdgcn_wmma_f32_16x16x32_f16(false, a, false, b, (short)0, c,
                                                false, false);
}

// ---------------------------------------------------------------------------
// Kernel 0: one streaming pass converting h [V,128] f32 -> f16 in workspace.
// h fits in L2 (25.6 MB); gathering it as f16 halves L2 gather bandwidth and
// removes all v_cvt work from the WMMA kernel's critical path.
// ---------------------------------------------------------------------------
__global__ __launch_bounds__(256) void cvt_h_kernel(
    const float* __restrict__ h, _Float16* __restrict__ h16, int n /* mult of 8 */)
{
  int i = (blockIdx.x * blockDim.x + threadIdx.x) * 8;
  for (; i < n; i += gridDim.x * blockDim.x * 8) {
    v4f a = *(const v4f*)(h + i);
    v4f b = *(const v4f*)(h + i + 4);
    v8h o;
    #pragma unroll
    for (int j = 0; j < 4; ++j) { o[j] = (_Float16)a[j]; o[4 + j] = (_Float16)b[j]; }
    *(v8h*)(h16 + i) = o;
  }
}

// ---------------------------------------------------------------------------
// Kernel 1: per-edge message MLP via WMMA + scatter-add (atomics)
//   X[e] = concat(h[src[e]], edge_attr[e])          [E,144] (pad->160)
//   msg  = relu(X W1 + b1) W2 + b2                  [E,128]
//   agg[dst] += msg ; deg[dst] += 1
// One wave handles a tile of 16 edges. 72 v_wmma per tile.
// USE_H16: gather pre-converted f16 rows (two global_load_b128 per K-chunk).
// ---------------------------------------------------------------------------
template <bool USE_H16>
__global__ __launch_bounds__(256) void edge_msg_kernel(
    const float* __restrict__ h, const _Float16* __restrict__ h16,
    const int* __restrict__ ei, const float* __restrict__ ea,
    const float* __restrict__ W1, const float* __restrict__ b1,
    const float* __restrict__ W2, const float* __restrict__ b2,
    float* __restrict__ agg, float* __restrict__ deg, int E)
{
  // Weights transposed to [n][k] in f16 so a B-fragment (one column, two
  // runs of 8 consecutive K) is two contiguous ds_load_b128 per lane.
  __shared__ __align__(16) _Float16 w1t[DFEAT][KPAD];
  __shared__ __align__(16) _Float16 w2t[DFEAT][DFEAT];
  __shared__ __align__(16) _Float16 ybuf[WAVES][16][DFEAT];

  const int tid = threadIdx.x;
  for (int i = tid; i < DFEAT * KPAD; i += blockDim.x) {
    int n = i / KPAD, k = i - n * KPAD;
    w1t[n][k] = (_Float16)((k < DFEAT + EFEAT) ? W1[k * DFEAT + n] : 0.0f);
  }
  for (int i = tid; i < DFEAT * DFEAT; i += blockDim.x) {
    int n = i >> 7, k = i & 127;
    w2t[n][k] = (_Float16)W2[k * DFEAT + n];
  }
  __syncthreads();

  const int wave = tid >> 5;
  const int lane = tid & 31;
  const int row  = lane & 15;   // A-matrix row (edge within tile)
  const int grp  = lane >> 4;   // lane half-group
  const int kb   = grp * 8;     // A layout: K base within a 32-chunk
  const int ncl  = lane & 15;   // C/D & B layout: N column

  const int ntiles = E >> 4;
  for (int tile = blockIdx.x * WAVES + wave; tile < ntiles;
       tile += gridDim.x * WAVES) {
    const int e0 = tile << 4;

    // ---- A fragments for GEMM1 (lane L: row=L&15, K = kb..kb+7 & kb+16..kb+23)
    const int s = ei[e0 + row];                 // src node index
    v16h a1[5];
    if (USE_H16) {
      const _Float16* hrow = h16 + (size_t)s * DFEAT;   // 256 B row, L2-resident
      #pragma unroll
      for (int c = 0; c < 4; ++c) {
        v8h lo = *(const v8h*)(hrow + c * 32 + kb);        // global_load_b128
        v8h hi = *(const v8h*)(hrow + c * 32 + 16 + kb);   // global_load_b128
        v16h a;
        #pragma unroll
        for (int i = 0; i < 8; ++i) { a[i] = lo[i]; a[8 + i] = hi[i]; }
        a1[c] = a;
      }
    } else {
      const float* hrow = h + (size_t)s * DFEAT;
      #pragma unroll
      for (int c = 0; c < 4; ++c) {
        v16h a;
        #pragma unroll
        for (int i = 0; i < 8; ++i) a[i]     = (_Float16)hrow[c * 32 + kb + i];
        #pragma unroll
        for (int i = 0; i < 8; ++i) a[8 + i] = (_Float16)hrow[c * 32 + 16 + kb + i];
        a1[c] = a;
      }
    }
    {   // chunk 4: K 128..143 = edge features, K 144..159 = zero pad
      const float* earow = ea + (size_t)(e0 + row) * EFEAT;
      v16h a;
      #pragma unroll
      for (int i = 0; i < 8; ++i) a[i]     = (_Float16)earow[kb + i];
      #pragma unroll
      for (int i = 0; i < 8; ++i) a[8 + i] = (_Float16)0.0f;
      a1[4] = a;
    }

    // ---- GEMM1: Y = relu(X W1 + b1) -> per-wave f16 tile in LDS
    #pragma unroll
    for (int j = 0; j < 8; ++j) {
      const int ncol = j * 16 + ncl;
      const float bias = b1[ncol];
      v8f acc;
      #pragma unroll
      for (int r = 0; r < 8; ++r) acc[r] = bias;
      #pragma unroll
      for (int c = 0; c < 5; ++c) {
        v16h b;   // lane: col=ncol, K = c*32 + grp*8 .. +7 & c*32 + 16 + grp*8 .. +7
        const _Float16* wc = &w1t[ncol][c * 32 + grp * 8];
        #pragma unroll
        for (int i = 0; i < 16; ++i) b[i] = wc[i < 8 ? i : 8 + i];
        acc = wmma_f16(a1[c], b, acc);
      }
      #pragma unroll
      for (int r = 0; r < 8; ++r) {   // C layout: M = r + 8*grp, N = ncl
        float v = acc[r] > 0.0f ? acc[r] : 0.0f;
        ybuf[wave][r + 8 * grp][ncol] = (_Float16)v;
      }
    }
    // order cross-lane LDS write -> read within this wave
    asm volatile("s_wait_dscnt 0" ::: "memory");

    // ---- A fragments for GEMM2 from LDS Y (contiguous 16B -> ds_load_b128)
    v16h a2[4];
    #pragma unroll
    for (int c = 0; c < 4; ++c) {
      v16h a;
      const _Float16* y0 = &ybuf[wave][row][c * 32 + kb];
      const _Float16* y1 = &ybuf[wave][row][c * 32 + 16 + kb];
      #pragma unroll
      for (int i = 0; i < 8; ++i) { a[i] = y0[i]; a[8 + i] = y1[i]; }
      a2[c] = a;
    }

    // dst node for each of this lane's 8 output rows (M = r + 8*grp)
    int drow[8];
    #pragma unroll
    for (int r = 0; r < 8; ++r) drow[r] = ei[E + e0 + r + 8 * grp];

    // ---- GEMM2 + scatter-add
    #pragma unroll
    for (int j = 0; j < 8; ++j) {
      const int ncol = j * 16 + ncl;
      const float bias = b2[ncol];
      v8f acc;
      #pragma unroll
      for (int r = 0; r < 8; ++r) acc[r] = bias;
      #pragma unroll
      for (int c = 0; c < 4; ++c) {
        v16h b;
        const _Float16* wc = &w2t[ncol][c * 32 + grp * 8];
        #pragma unroll
        for (int i = 0; i < 16; ++i) b[i] = wc[i < 8 ? i : 8 + i];
        acc = wmma_f16(a2[c], b, acc);
      }
      #pragma unroll
      for (int r = 0; r < 8; ++r)
        atomicAdd(&agg[(size_t)drow[r] * DFEAT + ncol], acc[r]);
    }
    // degree: one +1 per edge
    if (lane < 16) atomicAdd(&deg[ei[E + e0 + lane]], 1.0f);
  }
}

// ---------------------------------------------------------------------------
// Kernel 2: out = LayerNorm(h + agg/(1+deg)) * gamma + beta
// One wave32 per node row (D=128 -> 4 elems/lane), shfl_xor reductions.
// ---------------------------------------------------------------------------
__global__ __launch_bounds__(256) void finalize_kernel(
    const float* __restrict__ h, const float* __restrict__ agg,
    const float* __restrict__ deg, const float* __restrict__ gamma,
    const float* __restrict__ beta, float* __restrict__ out, int V)
{
  const int wave = threadIdx.x >> 5;
  const int lane = threadIdx.x & 31;
  const int v = blockIdx.x * (blockDim.x >> 5) + wave;
  if (v >= V) return;

  const float inv = 1.0f / (1.0f + deg[v]);
  const size_t base = (size_t)v * DFEAT;

  float x[4];
  float sum = 0.0f;
  #pragma unroll
  for (int i = 0; i < 4; ++i) {
    const int c = lane + 32 * i;
    x[i] = h[base + c] + agg[base + c] * inv;
    sum += x[i];
  }
  #pragma unroll
  for (int off = 16; off > 0; off >>= 1) sum += __shfl_xor(sum, off, 32);
  const float mu = sum * (1.0f / (float)DFEAT);

  float var = 0.0f;
  #pragma unroll
  for (int i = 0; i < 4; ++i) { float t = x[i] - mu; var += t * t; }
  #pragma unroll
  for (int off = 16; off > 0; off >>= 1) var += __shfl_xor(var, off, 32);
  const float rs = rsqrtf(var * (1.0f / (float)DFEAT) + 1e-5f);

  #pragma unroll
  for (int i = 0; i < 4; ++i) {
    const int c = lane + 32 * i;
    out[base + c] = (x[i] - mu) * rs * gamma[c] + beta[c];
  }
}

extern "C" void kernel_launch(void* const* d_in, const int* in_sizes, int n_in,
                              void* d_out, int out_size, void* d_ws, size_t ws_size,
                              hipStream_t stream) {
  const float* h     = (const float*)d_in[0];
  const int*   ei    = (const int*)d_in[1];   // jax default x64-off -> int32
  const float* ea    = (const float*)d_in[2];
  const float* W1    = (const float*)d_in[3];
  const float* b1    = (const float*)d_in[4];
  const float* W2    = (const float*)d_in[5];
  const float* b2    = (const float*)d_in[6];
  const float* gamma = (const float*)d_in[7];
  const float* beta  = (const float*)d_in[8];
  float* out = (float*)d_out;

  const int V = in_sizes[0] / DFEAT;
  const int E = in_sizes[2] / EFEAT;

  const size_t agg_elems = (size_t)V * DFEAT + V;          // agg + deg (f32)
  const size_t base_bytes = agg_elems * sizeof(float);
  const size_t h16_bytes = (size_t)V * DFEAT * sizeof(_Float16);

  float* agg = (float*)d_ws;                 // [V, 128]
  float* deg = agg + (size_t)V * DFEAT;      // [V]
  hipMemsetAsync(d_ws, 0, base_bytes, stream);

  const bool use_h16 = (ws_size >= base_bytes + h16_bytes);
  _Float16* h16 = (_Float16*)((char*)d_ws + base_bytes);

  if (use_h16) {
    cvt_h_kernel<<<512, 256, 0, stream>>>(h, h16, V * DFEAT);
    edge_msg_kernel<true><<<1024, 256, 0, stream>>>(h, h16, ei, ea, W1, b1,
                                                    W2, b2, agg, deg, E);
  } else {
    edge_msg_kernel<false><<<1024, 256, 0, stream>>>(h, nullptr, ei, ea, W1, b1,
                                                     W2, b2, agg, deg, E);
  }
  finalize_kernel<<<(V + WAVES - 1) / WAVES, 256, 0, stream>>>(
      h, agg, deg, gamma, beta, out, V);
}
